// CausalSelfAttentionAfterLN_87428354277595
// MI455X (gfx1250) — compile-verified, matrix operand order checked by reading, not run on hardware
//
#include <hip/hip_runtime.h>

// ---------------- problem constants ----------------
constexpr int Bq   = 4;
constexpr int T    = 1024;
constexpr int H    = 12;
constexpr int E    = 768;
constexpr int DH   = 64;
constexpr int E1   = E + 1;      // 769
constexpr int N3E  = 3 * E;      // 2304
constexpr int KPAD = 800;        // 769 padded to multiple of 32
constexpr int BT   = Bq * T;     // 4096
constexpr float LOG_MU = -6.9314718055994531f;  // -log(1024)

// ---------------- WMMA types ----------------
typedef __attribute__((ext_vector_type(16))) __bf16 v16bf;
typedef __attribute__((ext_vector_type(8)))  __bf16 bf8;
typedef __attribute__((ext_vector_type(8)))  float  v8f;

__device__ inline v8f wmma_bf16(v16bf a, v16bf b, v8f c) {
  return __builtin_amdgcn_wmma_f32_16x16x32_bf16(false, a, false, b,
                                                 (short)0, c, false, false);
}

// Load a 16-element bf16 A/B fragment slice for this lane.
// p = &M[row*stride + kbase + half*8]; elements 0..7 <- p[0..7], 8..15 <- p[16..23]
// (matches CDNA5 16-bit A-matrix 16x32 VGPR layout; B mirrored with n as "row")
__device__ inline v16bf ld_frag(const __bf16* p) {
  bf8 lo = *(const bf8*)p;
  bf8 hi = *(const bf8*)(p + 16);
  v16bf r;
#pragma unroll
  for (int i = 0; i < 8; ++i) { r[i] = lo[i]; r[i + 8] = hi[i]; }
  return r;
}

// ---------------- 1. LayerNorm -> bf16 (K padded with zeros) ----------------
__global__ __launch_bounds__(256) void k_ln(const float* __restrict__ x,
                                            const float* __restrict__ lw,
                                            const float* __restrict__ lb,
                                            __bf16* __restrict__ xnb) {
  int row = blockIdx.x;                       // BT rows
  const float* xr = x + (size_t)row * E1;
  __shared__ float red[256];
  float s = 0.f;
  for (int i = threadIdx.x; i < E1; i += 256) s += xr[i];
  red[threadIdx.x] = s; __syncthreads();
  for (int o = 128; o > 0; o >>= 1) {
    if (threadIdx.x < o) red[threadIdx.x] += red[threadIdx.x + o];
    __syncthreads();
  }
  float mean = red[0] / (float)E1;
  __syncthreads();
  float v = 0.f;
  for (int i = threadIdx.x; i < E1; i += 256) { float d = xr[i] - mean; v += d * d; }
  red[threadIdx.x] = v; __syncthreads();
  for (int o = 128; o > 0; o >>= 1) {
    if (threadIdx.x < o) red[threadIdx.x] += red[threadIdx.x + o];
    __syncthreads();
  }
  float rstd = rsqrtf(red[0] / (float)E1 + 1e-5f);
  __bf16* o = xnb + (size_t)row * KPAD;
  for (int i = threadIdx.x; i < KPAD; i += 256)
    o[i] = (i < E1) ? (__bf16)(((xr[i] - mean) * rstd) * lw[i] + lb[i]) : (__bf16)0.f;
}

// ---------------- 2. pack weights transposed to bf16 ----------------
__global__ void k_pack_wattn(const float* __restrict__ W, __bf16* __restrict__ wab) {
  int idx = blockIdx.x * 256 + threadIdx.x;   // over N3E*KPAD, wab[c][k]
  if (idx >= N3E * KPAD) return;
  int c = idx / KPAD, k = idx - c * KPAD;
  wab[idx] = (k < E1) ? (__bf16)W[(size_t)k * N3E + c] : (__bf16)0.f;
}

__global__ void k_pack_wproj(const float* __restrict__ W, __bf16* __restrict__ wpb) {
  int idx = blockIdx.x * 256 + threadIdx.x;   // over E*E, wpb[n][k]
  if (idx >= E * E) return;
  int n = idx / E, k = idx - n * E;
  wpb[idx] = (__bf16)W[(size_t)k * E + n];
}

// ---------------- 3. QKV GEMM (bf16 WMMA), scatter to q/k/v layouts --------
// q,k: [B,H,T,DH] bf16 (q pre-scaled by 1/8);  v: [B,H,DH,T] bf16 (transposed)
__global__ __launch_bounds__(128) void k_qkv(const __bf16* __restrict__ xnb,
                                             const __bf16* __restrict__ wab,
                                             const float* __restrict__ bias,
                                             __bf16* __restrict__ qb,
                                             __bf16* __restrict__ kb,
                                             __bf16* __restrict__ vb) {
  int wid = threadIdx.x >> 5, lane = threadIdx.x & 31;
  int half = lane >> 4, l16 = lane & 15;
  int mb = blockIdx.x * 64 + wid * 16;        // grid.x = 64
  int nb = blockIdx.y * 64;                   // grid.y = 36
  v8f acc[4] = {};
  for (int k = 0; k < KPAD; k += 32) {
    v16bf a = ld_frag(xnb + (size_t)(mb + l16) * KPAD + k + half * 8);
#pragma unroll
    for (int j = 0; j < 4; ++j) {
      v16bf b = ld_frag(wab + (size_t)(nb + j * 16 + l16) * KPAD + k + half * 8);
      acc[j] = wmma_bf16(a, b, acc[j]);
    }
  }
#pragma unroll
  for (int j = 0; j < 4; ++j) {
#pragma unroll
    for (int i = 0; i < 8; ++i) {
      int m = mb + i + half * 8;
      int c = nb + j * 16 + l16;
      float val = acc[j][i] + bias[c];
      int sec = (c >= 2 * E) ? 2 : (c >= E ? 1 : 0);
      int e = c - sec * E;
      int h = e >> 6, d = e & 63;
      int bb = m >> 10, t = m & 1023;
      size_t bh = (size_t)bb * H + h;
      if (sec == 0)      qb[(bh * T + t) * DH + d] = (__bf16)(val * 0.125f);
      else if (sec == 1) kb[(bh * T + t) * DH + d] = (__bf16)val;
      else               vb[(bh * DH + d) * T + t] = (__bf16)val;
    }
  }
}

// ---------------- 4. attention raw scores S = q @ k^T ----------------
__global__ __launch_bounds__(128) void k_scores(const __bf16* __restrict__ qb,
                                                const __bf16* __restrict__ kb,
                                                float* __restrict__ att) {
  int bhr = blockIdx.x;                       // B*H*(T/64)
  int bh = bhr >> 4;
  int mb0 = (bhr & 15) * 64;
  int nb0 = blockIdx.y * 64;                  // grid.y = 16
  if (nb0 > mb0 + 63) return;                 // fully above diagonal -> masked
  int wid = threadIdx.x >> 5, lane = threadIdx.x & 31;
  int half = lane >> 4, l16 = lane & 15;
  int mb = mb0 + wid * 16;
  const __bf16* q  = qb + (size_t)bh * T * DH;
  const __bf16* kk = kb + (size_t)bh * T * DH;
  float* A = att + (size_t)bh * T * T;
  v8f acc[4] = {};
  for (int k = 0; k < DH; k += 32) {
    v16bf a = ld_frag(q + (size_t)(mb + l16) * DH + k + half * 8);
#pragma unroll
    for (int j = 0; j < 4; ++j) {
      v16bf b = ld_frag(kk + (size_t)(nb0 + j * 16 + l16) * DH + k + half * 8);
      acc[j] = wmma_bf16(a, b, acc[j]);
    }
  }
#pragma unroll
  for (int j = 0; j < 4; ++j)
#pragma unroll
    for (int i = 0; i < 8; ++i)
      A[(size_t)(mb + i + half * 8) * T + nb0 + j * 16 + l16] = acc[j][i];
}

// ---------------- 5. causal softmax in place (one wave per row) ------------
__global__ __launch_bounds__(256) void k_softmax(float* __restrict__ att) {
  int gid = blockIdx.x * 8 + (threadIdx.x >> 5);   // B*H*T rows total
  int lane = threadIdx.x & 31;
  int bh = gid >> 10, m = gid & 1023;
  float* row = att + ((size_t)bh * T + m) * T;
  float mx = -__builtin_inff();
  for (int c = lane; c <= m; c += 32) mx = fmaxf(mx, row[c]);
#pragma unroll
  for (int o = 1; o < 32; o <<= 1) mx = fmaxf(mx, __shfl_xor(mx, o, 32));
  float s = 0.f;
  for (int c = lane; c <= m; c += 32) s += __expf(row[c] - mx);
#pragma unroll
  for (int o = 1; o < 32; o <<= 1) s += __shfl_xor(s, o, 32);
  float inv = 1.f / s;
  for (int c = lane; c < T; c += 32)
    row[c] = (c <= m) ? __expf(row[c] - mx) * inv : 0.f;
}

// ---------------- 6. Sinkhorn ----------------
__global__ void k_zero(float* __restrict__ p, int n) {
  int i = blockIdx.x * 256 + threadIdx.x;
  if (i < n) p[i] = 0.f;
}

// f[q] = log_mu - lse_k( g[k] - cost[q,k] )   (one wave per row)
__global__ __launch_bounds__(256) void k_sink_f(const float* __restrict__ att,
                                                const float* __restrict__ g,
                                                float* __restrict__ f) {
  int gid = blockIdx.x * 8 + (threadIdx.x >> 5);
  int lane = threadIdx.x & 31;
  int bh = gid >> 10, q = gid & 1023;
  const float* row = att + ((size_t)bh * T + q) * T;
  const float* gv = g + (size_t)bh * T;
  float m = -__builtin_inff(), s = 0.f;
  for (int c = lane; c < T; c += 32) {
    float v = gv[c] - row[c];
    if (v > m) { s = s * __expf(m - v) + 1.f; m = v; }
    else       { s += __expf(v - m); }
  }
#pragma unroll
  for (int o = 1; o < 32; o <<= 1) {
    float m2 = __shfl_xor(m, o, 32);
    float s2 = __shfl_xor(s, o, 32);
    float mn = fmaxf(m, m2);
    s = s * __expf(m - mn) + s2 * __expf(m2 - mn);
    m = mn;
  }
  if (lane == 0) f[(size_t)bh * T + q] = LOG_MU - (m + __logf(s));
}

// g[k] = log_mu - lse_q( f[q] - cost[q,k] )   (one thread per column)
__global__ __launch_bounds__(256) void k_sink_g(const float* __restrict__ att,
                                                const float* __restrict__ f,
                                                float* __restrict__ g) {
  int gid = blockIdx.x * 256 + threadIdx.x;   // B*H*T columns total
  int bh = gid >> 10, c = gid & 1023;
  const float* A = att + (size_t)bh * T * T;
  const float* fv = f + (size_t)bh * T;
  float m = -__builtin_inff(), s = 0.f;
  for (int q = 0; q < T; ++q) {
    float v = fv[q] - A[(size_t)q * T + c];
    if (v > m) { s = s * __expf(m - v) + 1.f; m = v; }
    else       { s += __expf(v - m); }
  }
  g[gid] = LOG_MU - (m + __logf(s));
}

// ---------------- 7. y = pi @ v  (pi = exp(f+g-cost)*T built on the fly) ---
// vb is [B,H,DH,T]; output yb is [B,T,E] bf16 (heads interleaved back)
__global__ __launch_bounds__(128) void k_pv(const float* __restrict__ att,
                                            const float* __restrict__ f,
                                            const float* __restrict__ g,
                                            const __bf16* __restrict__ vb,
                                            __bf16* __restrict__ yb) {
  int blk = blockIdx.x;                       // B*H*(T/64)
  int bh = blk >> 4;
  int mb = (blk & 15) * 64 + (threadIdx.x >> 5) * 16;
  int lane = threadIdx.x & 31, half = lane >> 4, l16 = lane & 15;
  const float* A  = att + (size_t)bh * T * T;
  const float* fv = f + (size_t)bh * T;
  const float* gv = g + (size_t)bh * T;
  const __bf16* v = vb + (size_t)bh * DH * T;
  int b = bh / H, h = bh - b * H;
  int r = mb + l16;
  float fr = fv[r];
  v8f acc[4] = {};
  for (int k = 0; k < T; k += 32) {
    const float* pc = A + (size_t)r * T + k + half * 8;
    const float* pg = gv + k + half * 8;
    v16bf a;
#pragma unroll
    for (int j = 0; j < 8; ++j) {
      a[j]     = (__bf16)(__expf(fr + pg[j]      - pc[j])      * 1024.0f);
      a[j + 8] = (__bf16)(__expf(fr + pg[j + 16] - pc[j + 16]) * 1024.0f);
    }
#pragma unroll
    for (int j = 0; j < 4; ++j) {
      v16bf bfr = ld_frag(v + (size_t)(j * 16 + l16) * T + k + half * 8);
      acc[j] = wmma_bf16(a, bfr, acc[j]);
    }
  }
#pragma unroll
  for (int j = 0; j < 4; ++j)
#pragma unroll
    for (int i = 0; i < 8; ++i) {
      int t = mb + i + half * 8;
      int d = j * 16 + l16;
      yb[((size_t)b * T + t) * E + h * DH + d] = (__bf16)acc[j][i];
    }
}

// ---------------- 8. output projection ----------------
__global__ __launch_bounds__(128) void k_proj(const __bf16* __restrict__ yb,
                                              const __bf16* __restrict__ wpb,
                                              const float* __restrict__ bias,
                                              float* __restrict__ out) {
  int wid = threadIdx.x >> 5, lane = threadIdx.x & 31;
  int half = lane >> 4, l16 = lane & 15;
  int mb = blockIdx.x * 64 + wid * 16;        // grid.x = 64
  int nb = blockIdx.y * 64;                   // grid.y = 12
  v8f acc[4] = {};
  for (int k = 0; k < E; k += 32) {
    v16bf a = ld_frag(yb + (size_t)(mb + l16) * E + k + half * 8);
#pragma unroll
    for (int j = 0; j < 4; ++j) {
      v16bf b = ld_frag(wpb + (size_t)(nb + j * 16 + l16) * E + k + half * 8);
      acc[j] = wmma_bf16(a, b, acc[j]);
    }
  }
#pragma unroll
  for (int j = 0; j < 4; ++j)
#pragma unroll
    for (int i = 0; i < 8; ++i) {
      int m = mb + i + half * 8;
      int n = nb + j * 16 + l16;
      out[(size_t)m * E + n] = acc[j][i] + bias[n];
    }
}

// ---------------- host launcher ----------------
extern "C" void kernel_launch(void* const* d_in, const int* in_sizes, int n_in,
                              void* d_out, int out_size, void* d_ws, size_t ws_size,
                              hipStream_t stream) {
  const float* x      = (const float*)d_in[0];
  const float* ln_w   = (const float*)d_in[1];
  const float* ln_b   = (const float*)d_in[2];
  const float* W_attn = (const float*)d_in[3];
  const float* b_attn = (const float*)d_in[4];
  const float* W_proj = (const float*)d_in[5];
  const float* b_proj = (const float*)d_in[6];
  float* out = (float*)d_out;

  char* w = (char*)d_ws;
  auto carve = [&](size_t bytes) {
    char* p = w;
    w += (bytes + 255) & ~(size_t)255;
    return p;
  };
  __bf16* xnb = (__bf16*)carve((size_t)BT * KPAD * 2);
  __bf16* wab = (__bf16*)carve((size_t)N3E * KPAD * 2);
  __bf16* wpb = (__bf16*)carve((size_t)E * E * 2);
  __bf16* qb  = (__bf16*)carve((size_t)Bq * H * T * DH * 2);
  __bf16* kb  = (__bf16*)carve((size_t)Bq * H * T * DH * 2);
  __bf16* vb  = (__bf16*)carve((size_t)Bq * H * T * DH * 2);
  float*  att = (float*) carve((size_t)Bq * H * T * T * 4);
  float*  fp  = (float*) carve((size_t)Bq * H * T * 4);
  float*  gp  = (float*) carve((size_t)Bq * H * T * 4);
  __bf16* yb  = (__bf16*)carve((size_t)BT * E * 2);

  // 1. LayerNorm + bf16 pack
  k_ln<<<BT, 256, 0, stream>>>(x, ln_w, ln_b, xnb);
  // 2. weight packs (transposed, padded)
  k_pack_wattn<<<(N3E * KPAD + 255) / 256, 256, 0, stream>>>(W_attn, wab);
  k_pack_wproj<<<(E * E + 255) / 256, 256, 0, stream>>>(W_proj, wpb);
  // 3. QKV GEMM
  k_qkv<<<dim3(BT / 64, N3E / 64), 128, 0, stream>>>(xnb, wab, b_attn, qb, kb, vb);
  // 4. scores + 5. causal softmax
  k_scores<<<dim3(Bq * H * (T / 64), T / 64), 128, 0, stream>>>(qb, kb, att);
  k_softmax<<<Bq * H * T / 8, 256, 0, stream>>>(att);
  // 6. Sinkhorn: g=0, then 6 x (f-update, g-update)
  k_zero<<<(Bq * H * T + 255) / 256, 256, 0, stream>>>(gp, Bq * H * T);
  for (int it = 0; it < 6; ++it) {
    k_sink_f<<<Bq * H * T / 8, 256, 0, stream>>>(att, gp, fp);
    k_sink_g<<<Bq * H * T / 256, 256, 0, stream>>>(att, fp, gp);
  }
  // 7. y = pi @ v
  k_pv<<<Bq * H * (T / 64), 128, 0, stream>>>(att, fp, gp, vb, yb);
  // 8. projection
  k_proj<<<dim3(BT / 64, E / 64), 128, 0, stream>>>(yb, wpb, b_proj, out);
}
